// HausdorffLoss_6863357739536
// MI455X (gfx1250) — matrix operand order repaired and checked by reading, not compiled, not run on hardware
//
#include <hip/hip_runtime.h>
#include <hip/hip_bf16.h>

// Hausdorff loss: B=8, N=M=4096, D=64, fp32.
// dist^2(n,m) = ||x_n||^2 + ||y_m||^2 - 2 x_n.y_m  via split-f16 WMMA gram.
// out = sqrt(max_bn min_m d2) + sqrt(max_bm min_n d2)

#define B_  8
#define N_  4096
#define M_  4096
#define D_  64
#define TS  128        // tile size (n and m)
#define RH  72         // LDS row stride in halves (64 + 8 pad -> conflict-free b128)

typedef __attribute__((ext_vector_type(16))) _Float16 v16h;
typedef __attribute__((ext_vector_type(8)))  _Float16 v8h;
typedef __attribute__((ext_vector_type(8)))  float    v8f;

#define INF_BITS 0x7F800000u

// ---------------- kernel 0: init min buffers + scalars ----------------
__global__ void hd_init(unsigned* __restrict__ ws) {
    int i = blockIdx.x * 256 + threadIdx.x;
    if (i < B_ * N_ + B_ * M_) ws[i] = INF_BITS;      // rowmin + colmin
    else if (i < B_ * N_ + B_ * M_ + 2) ws[i] = 0u;   // max scalars
}

// ---------------- kernel 1: squared row norms ----------------
__global__ void hd_norms(const float* __restrict__ X, const float* __restrict__ Y,
                         float* __restrict__ xn, float* __restrict__ yn) {
    int i = blockIdx.x * 256 + threadIdx.x;           // 0 .. 2*B*N-1
    const float* src; float* dst; int j;
    if (i < B_ * N_) { src = X; dst = xn; j = i; }
    else             { src = Y; dst = yn; j = i - B_ * N_; }
    const float4* p = (const float4*)(src + (size_t)j * D_);
    float s = 0.f;
#pragma unroll
    for (int c = 0; c < D_ / 4; ++c) {
        float4 v = p[c];
        s += v.x * v.x + v.y * v.y + v.z * v.z + v.w * v.w;
    }
    dst[j] = s;
}

// ---------------- fragment helpers ----------------
__device__ __forceinline__ v16h load_frag(const _Float16* tile, int row, int k0, int k1) {
    union { v16h v; v8h h[2]; } u;
    u.h[0] = *(const v8h*)(tile + row * RH + k0);
    u.h[1] = *(const v8h*)(tile + row * RH + k1);
    return u.v;
}

__device__ __forceinline__ void split_store(_Float16* hi, _Float16* lo, int o, float v) {
    _Float16 h = (_Float16)v;
    hi[o] = h;
    lo[o] = (_Float16)(v - (float)h);
}

// ---------------- kernel 2: tiled gram + tile min reduction ----------------
__global__ __launch_bounds__(256)
void hd_tile(const float* __restrict__ X, const float* __restrict__ Y,
             const float* __restrict__ xn, const float* __restrict__ yn,
             unsigned* __restrict__ rowmin, unsigned* __restrict__ colmin) {
    __shared__ __align__(16) _Float16 lAhi[TS * RH];
    __shared__ __align__(16) _Float16 lAlo[TS * RH];
    __shared__ __align__(16) _Float16 lBhi[TS * RH];
    __shared__ __align__(16) _Float16 lBlo[TS * RH];
    __shared__ float    lXn[TS], lYn[TS];
    __shared__ unsigned lRow[TS], lCol[TS];

    const int t  = threadIdx.x;
    const int b  = blockIdx.z;
    const int n0 = blockIdx.y * TS;
    const int m0 = blockIdx.x * TS;

    // wave-uniform guarded init (waves 0-3 vs 4-7; EXEC stays full per wave)
    if (t < TS) {
        lRow[t] = INF_BITS; lCol[t] = INF_BITS;
        lXn[t]  = xn[b * N_ + n0 + t];
    } else {
        lYn[t - TS] = yn[b * M_ + m0 + (t - TS)];
    }

    // load 128x64 fp32 tiles, split into f16 hi/lo in LDS
    const float* Xb = X + ((size_t)b * N_ + n0) * D_;
    const float* Yb = Y + ((size_t)b * M_ + m0) * D_;
#pragma unroll
    for (int i = 0; i < 8; ++i) {
        int f   = t + i * 256;          // 0..2047 float4 slots
        int row = f >> 4;
        int c4  = (f & 15) * 4;
        float4 vx = *(const float4*)(Xb + row * D_ + c4);
        float4 vy = *(const float4*)(Yb + row * D_ + c4);
        int o = row * RH + c4;
        split_store(lAhi, lAlo, o + 0, vx.x);
        split_store(lAhi, lAlo, o + 1, vx.y);
        split_store(lAhi, lAlo, o + 2, vx.z);
        split_store(lAhi, lAlo, o + 3, vx.w);
        split_store(lBhi, lBlo, o + 0, vy.x);
        split_store(lBhi, lBlo, o + 1, vy.y);
        split_store(lBhi, lBlo, o + 2, vy.z);
        split_store(lBhi, lBlo, o + 3, vy.w);
    }
    __syncthreads();

    const int lane = t & 31;
    const int w    = t >> 5;        // wave 0..7 -> n strip
    const int hi   = lane >> 4;     // half of wave
    const int l15  = lane & 15;

    v8f acc[8] = {};                // 8 subtiles of 16x16 along m

    const int arow = w * 16 + l15;  // A-fragment row (M index) for this lane
#pragma unroll
    for (int ks = 0; ks < 2; ++ks) {
        const int kb = ks * 32;
        // A frag: halves 0-15 = K kb+16*hi .. kb+16*hi+15 (contiguous)
        v16h ahi = load_frag(lAhi, arow, kb + 16 * hi, kb + 16 * hi + 8);
        v16h alo = load_frag(lAlo, arow, kb + 16 * hi, kb + 16 * hi + 8);
        // term hi*hi
#pragma unroll
        for (int mb = 0; mb < 8; ++mb) {
            v16h bh = load_frag(lBhi, mb * 16 + l15, kb + 16 * hi, kb + 16 * hi + 8);
            acc[mb] = __builtin_amdgcn_wmma_f32_16x16x32_f16(
                false, ahi, false, bh, (short)0, acc[mb], false, false);
        }
        // term hi*lo
#pragma unroll
        for (int mb = 0; mb < 8; ++mb) {
            v16h bl = load_frag(lBlo, mb * 16 + l15, kb + 16 * hi, kb + 16 * hi + 8);
            acc[mb] = __builtin_amdgcn_wmma_f32_16x16x32_f16(
                false, ahi, false, bl, (short)0, acc[mb], false, false);
        }
        // term lo*hi
#pragma unroll
        for (int mb = 0; mb < 8; ++mb) {
            v16h bh = load_frag(lBhi, mb * 16 + l15, kb + 16 * hi, kb + 16 * hi + 8);
            acc[mb] = __builtin_amdgcn_wmma_f32_16x16x32_f16(
                false, alo, false, bh, (short)0, acc[mb], false, false);
        }
    }

    // d2 = xn + yn - 2*gram, clamp; then tile-local row/col mins.
    float xv[8];
#pragma unroll
    for (int j = 0; j < 8; ++j) xv[j] = lXn[w * 16 + 8 * hi + j];
    float yv[8];
#pragma unroll
    for (int mb = 0; mb < 8; ++mb) yv[mb] = lYn[mb * 16 + l15];

    float pm[8];                         // per-n partial min over this lane's m's
#pragma unroll
    for (int j = 0; j < 8; ++j) pm[j] = __uint_as_float(INF_BITS);
    float cmin[8];                       // per-m-subtile min over this lane's n's
#pragma unroll
    for (int mb = 0; mb < 8; ++mb) {
        float cm = __uint_as_float(INF_BITS);
#pragma unroll
        for (int j = 0; j < 8; ++j) {
            float sq = fmaxf(xv[j] + yv[mb] - 2.0f * acc[mb][j], 1e-12f);
            cm    = fminf(cm, sq);
            pm[j] = fminf(pm[j], sq);
        }
        cmin[mb] = cm;
    }

    // column mins: combine the two lane-halves (same m, different n ranges)
#pragma unroll
    for (int mb = 0; mb < 8; ++mb) {
        float c = fminf(cmin[mb], __shfl_xor(cmin[mb], 16, 32));
        if (hi == 0)
            atomicMin(&lCol[mb * 16 + l15], __float_as_uint(c));
    }
    // row mins: reduce across the 16 lanes of each half
#pragma unroll
    for (int j = 0; j < 8; ++j) {
        float r = pm[j];
        r = fminf(r, __shfl_xor(r, 1, 32));
        r = fminf(r, __shfl_xor(r, 2, 32));
        r = fminf(r, __shfl_xor(r, 4, 32));
        r = fminf(r, __shfl_xor(r, 8, 32));
        if (l15 == 0)
            atomicMin(&lRow[w * 16 + 8 * hi + j], __float_as_uint(r));
    }
    __syncthreads();

    if (t < TS) atomicMin(&rowmin[b * N_ + n0 + t], lRow[t]);
    else        atomicMin(&colmin[b * M_ + m0 + (t - TS)], lCol[t - TS]);
}

// ---------------- kernel 3: global max of the min arrays ----------------
__global__ void hd_maxreduce(const unsigned* __restrict__ mins,
                             unsigned* __restrict__ scal) {
    __shared__ unsigned sm[256];
    int tid = threadIdx.x;
    int i   = blockIdx.x * 256 + tid;
    sm[tid] = mins[i];
    __syncthreads();
#pragma unroll
    for (int s = 128; s > 0; s >>= 1) {
        if (tid < s) sm[tid] = max(sm[tid], sm[tid + s]);
        __syncthreads();
    }
    if (tid == 0) {
        // blocks 0..127 cover rowmin, 128..255 cover colmin
        atomicMax(&scal[blockIdx.x < (B_ * N_) / 256 ? 0 : 1], sm[0]);
    }
}

// ---------------- kernel 4: final scalar ----------------
__global__ void hd_final(const unsigned* __restrict__ scal, float* __restrict__ out) {
    float a = __uint_as_float(scal[0]);
    float b = __uint_as_float(scal[1]);
    out[0] = sqrtf(fmaxf(a, 1e-12f)) + sqrtf(fmaxf(b, 1e-12f));
}

extern "C" void kernel_launch(void* const* d_in, const int* in_sizes, int n_in,
                              void* d_out, int out_size, void* d_ws, size_t ws_size,
                              hipStream_t stream) {
    (void)in_sizes; (void)n_in; (void)out_size; (void)ws_size;
    const float* pred  = (const float*)d_in[0];
    const float* label = (const float*)d_in[1];
    float* out = (float*)d_out;

    // workspace layout (uints): [rowmin BN][colmin BM][2 max scalars][xn BN f32][yn BM f32]
    unsigned* ws      = (unsigned*)d_ws;
    unsigned* rowmin  = ws;
    unsigned* colmin  = ws + B_ * N_;
    unsigned* scal    = ws + B_ * N_ + B_ * M_;
    float*    xnorm   = (float*)(scal + 2);
    float*    ynorm   = xnorm + B_ * N_;

    const int ninit = B_ * N_ + B_ * M_ + 2;
    hd_init<<<(ninit + 255) / 256, 256, 0, stream>>>(ws);
    hd_norms<<<(2 * B_ * N_) / 256, 256, 0, stream>>>(pred, label, xnorm, ynorm);

    dim3 grid(M_ / TS, N_ / TS, B_);
    hd_tile<<<grid, 256, 0, stream>>>(pred, label, xnorm, ynorm, rowmin, colmin);

    hd_maxreduce<<<(B_ * N_ + B_ * M_) / 256, 256, 0, stream>>>(ws, scal);
    hd_final<<<1, 1, 0, stream>>>(scal, out);
}
